// torch_kmeans_68023692034289
// MI455X (gfx1250) — compile-verified, matrix operand order checked by reading, not run on hardware
//
#include <hip/hip_runtime.h>

// ---------------------------------------------------------------------------
// K-means (Lloyd) for MI455X / gfx1250: wave32, WMMA bf16, TDM-staged B tiles,
// counting-sort update.
// d_in[0] = X           [400000 x 128] f32
// d_in[1] = init_cent   [512 x 128]    f32
// d_out   = [centroids 512*128 f32][labels 400000 int32 bits][inertia 1 f32]
// ---------------------------------------------------------------------------

typedef __attribute__((ext_vector_type(16))) __bf16 v16bf;
typedef __attribute__((ext_vector_type(8)))  __bf16 v8bf;
typedef __attribute__((ext_vector_type(8)))  float  v8f;
typedef __attribute__((ext_vector_type(4)))  float  f4;
typedef __attribute__((ext_vector_type(4)))  unsigned u32x4;
typedef __attribute__((ext_vector_type(8)))  int      i32x8;
typedef __attribute__((ext_vector_type(4)))  int      i32x4;

#define KM_N 400000
#define KM_D 128
#define KM_K 512
#define KM_ITERS 10
#define KM_NT (KM_K / 16)  // 32 cluster tiles

#if __has_builtin(__builtin_amdgcn_tensor_load_to_lds) && \
    __has_builtin(__builtin_amdgcn_s_wait_tensorcnt)
#define HAS_TDM 1
#else
#define HAS_TDM 0
#endif

// Truncation split: v = hi + rem (rem exact by Sterbenz), rem truncated to bf16.
__device__ __forceinline__ __bf16 bf_hi(float v, float& rem) {
  unsigned u = __builtin_bit_cast(unsigned, v);
  float fh = __builtin_bit_cast(float, u & 0xffff0000u);
  rem = v - fh;
  return __builtin_bit_cast(__bf16, (unsigned short)(u >> 16));
}
__device__ __forceinline__ __bf16 bf_of(float v) {
  unsigned u = __builtin_bit_cast(unsigned, v);
  return __builtin_bit_cast(__bf16, (unsigned short)(u >> 16));
}

#if HAS_TDM
// TDM: DMA one [16 x 128] bf16 tile (row-major, 4KB) global -> LDS.
// D# fields per CDNA5 ISA.
__device__ __forceinline__ void tdm_load_tile(const __bf16* gsrc,
                                              unsigned lds_byte_addr) {
  unsigned long long ga = (unsigned long long)(const void*)gsrc;
  u32x4 g0;
  g0.x = 1u;                                    // count=1, user mode, no gather
  g0.y = lds_byte_addr;                         // lds_addr (bytes)
  g0.z = (unsigned)(ga & 0xffffffffu);          // global_addr[31:0]
  g0.w = (unsigned)((ga >> 32) & 0x01ffffffu)   // global_addr[56:32]
         | (2u << 30);                          // type = 2 ("image")
  i32x8 g1 = {};
  g1[0] = (1 << 16);          // data_size=1 (2 bytes); wg_mask=0; no pad/iter
  g1[1] = (KM_D << 16);       // tensor_dim0[15:0]=128 in bits[31:16]
  g1[2] = (16 << 16);         // tensor_dim0 hi=0 | tensor_dim1[15:0]=16 (tile rows)
  g1[3] = (KM_D << 16);       // tensor_dim1 hi=0 | tile_dim0=128
  g1[4] = 16;                 // tile_dim1=16, tile_dim2=0
  g1[5] = KM_D;               // tensor_dim0_stride = 128 elements
  g1[6] = 0;
  g1[7] = 0;
  i32x4 gz = {};
#if __clang_major__ >= 23
  i32x8 gz8 = {};
  __builtin_amdgcn_tensor_load_to_lds(g0, g1, gz, gz, gz8, 0);
#else
  __builtin_amdgcn_tensor_load_to_lds(g0, g1, gz, gz, 0);
#endif
}
#endif

// ---------------------------------------------------------------------------
// ||x||^2 per sample: one wave per row.
// ---------------------------------------------------------------------------
__global__ __launch_bounds__(256) void kmeans_xsq(const float* __restrict__ X,
                                                  float* __restrict__ xsq) {
  const int lane = threadIdx.x & 31;
  const int wv   = threadIdx.x >> 5;
  const int row  = blockIdx.x * 8 + wv;
  f4 a = __builtin_nontemporal_load((const f4*)(X + (size_t)row * KM_D + lane * 4));
  float s = a.x * a.x + a.y * a.y + a.z * a.z + a.w * a.w;
#pragma unroll
  for (int m = 16; m >= 1; m >>= 1) s += __shfl_xor(s, m, 32);
  if (lane == 0) xsq[row] = s;
}

// ---------------------------------------------------------------------------
__global__ __launch_bounds__(256) void kmeans_zero(int* __restrict__ counts,
                                                   int* __restrict__ cursor) {
  const int i = blockIdx.x * 256 + threadIdx.x;
  if (i < KM_K) { counts[i] = 0; cursor[i] = 0; }
}

// ---------------------------------------------------------------------------
// Finishing: store centroid, bf16 hi/lo split, ||c||^2. Block=128, 1 blk/cluster.
// ---------------------------------------------------------------------------
__device__ __forceinline__ void finish_centroid(
    int k, int f, float c, float* __restrict__ cent, __bf16* __restrict__ Chi,
    __bf16* __restrict__ Clo, float* __restrict__ csq,
    float* __restrict__ outCent, int writeOut) {
  cent[(size_t)k * KM_D + f] = c;
  float rem;
  Chi[(size_t)k * KM_D + f] = bf_hi(c, rem);
  Clo[(size_t)k * KM_D + f] = bf_of(rem);

  float ss = c * c;
#pragma unroll
  for (int m = 16; m >= 1; m >>= 1) ss += __shfl_xor(ss, m, 32);
  __shared__ float wsum[4];
  if ((f & 31) == 0) wsum[f >> 5] = ss;
  __syncthreads();
  if (f == 0) csq[k] = wsum[0] + wsum[1] + wsum[2] + wsum[3];

  if (writeOut) outCent[(size_t)k * KM_D + f] = c;
}

__global__ __launch_bounds__(128) void kmeans_centinit(
    const float* __restrict__ src, float* __restrict__ cent,
    __bf16* __restrict__ Chi, __bf16* __restrict__ Clo, float* __restrict__ csq) {
  const int k = blockIdx.x, f = threadIdx.x;
  float c = src[(size_t)k * KM_D + f];
  finish_centroid(k, f, c, cent, Chi, Clo, csq, nullptr, 0);
}

// ---------------------------------------------------------------------------
// Assignment: one wave = 16 samples vs all 512 clusters.
// B tiles DMA'd by the Tensor Data Mover into double-buffered LDS (prefetch
// depth 1); two independent WMMA accumulator chains + ks-level B-fragment
// double buffer for XDL/LDS latency hiding.
// Tracks e = 0.5*||c||^2 - x.c; d^2 = ||x||^2 + 2e.
// ---------------------------------------------------------------------------
__global__ __launch_bounds__(256) void kmeans_assign(
    const float* __restrict__ X, const float* __restrict__ xsq,
    const __bf16* __restrict__ Chi, const __bf16* __restrict__ Clo,
    const float* __restrict__ csq,
    int* __restrict__ labels, int* __restrict__ counts,
    float* __restrict__ iblk) {
  const int lane    = threadIdx.x & 31;
  const int wv      = threadIdx.x >> 5;
  const int rowbase = blockIdx.x * 128 + wv * 16;
  const int col     = lane & 15;
  const int half    = lane >> 4;
  // Wave index as a compiler-known scalar -> pure s_cbranch leader checks
  // (no EXEC save/restore around the TDM issue in the hot loop).
  const int wvs = __builtin_amdgcn_readfirstlane(wv);

#if HAS_TDM
  __shared__ __align__(16) __bf16 BhS[2][16 * KM_D];
  __shared__ __align__(16) __bf16 BlS[2][16 * KM_D];
  // Kick off tile 0 DMA before the (long) A-fragment load/convert.
  if (wvs == 0) {
    tdm_load_tile(Chi, (unsigned)(size_t)(void*)&BhS[0][0]);
    tdm_load_tile(Clo, (unsigned)(size_t)(void*)&BlS[0][0]);
  }
#endif

  // ---- Load + split A fragments (X read exactly once per iteration) ----
  v16bf Ah[4], Al[4];
  const float* xrow = X + (size_t)(rowbase + col) * KM_D;
#pragma unroll
  for (int ks = 0; ks < 4; ++ks) {
    const float* p = xrow + ks * 32 + half * 8;
    f4 f0 = __builtin_nontemporal_load((const f4*)(p));
    f4 f1 = __builtin_nontemporal_load((const f4*)(p + 4));
    f4 f2 = __builtin_nontemporal_load((const f4*)(p + 16));
    f4 f3 = __builtin_nontemporal_load((const f4*)(p + 20));
    float vals[16] = {f0.x, f0.y, f0.z, f0.w, f1.x, f1.y, f1.z, f1.w,
                      f2.x, f2.y, f2.z, f2.w, f3.x, f3.y, f3.z, f3.w};
#pragma unroll
    for (int e = 0; e < 16; ++e) {
      float rem;
      Ah[ks][e] = bf_hi(vals[e], rem);
      Al[ks][e] = bf_of(rem);
    }
  }

  f4 q0 = *(const f4*)(xsq + rowbase + half * 8);
  f4 q1 = *(const f4*)(xsq + rowbase + half * 8 + 4);
  float xs[8] = {q0.x, q0.y, q0.z, q0.w, q1.x, q1.y, q1.z, q1.w};

  float beste[8];
  int   bestn[8];
#pragma unroll
  for (int v = 0; v < 8; ++v) { beste[v] = 3.4e38f; bestn[v] = 0; }

  union BU { v16bf v; v8bf h[2]; };

#pragma unroll 2  // makes nt&1 buffer selects compile-time constants
  for (int nt = 0; nt < KM_NT; ++nt) {
    const int cl = nt * 16 + col;
#if HAS_TDM
    if (wvs == 0) __builtin_amdgcn_s_wait_tensorcnt(0);  // tile nt landed
    __syncthreads();  // tile nt visible; buffer (nt+1)&1 no longer being read
    if (wvs == 0 && nt + 1 < KM_NT) {
      tdm_load_tile(Chi + (size_t)(nt + 1) * 16 * KM_D,
                    (unsigned)(size_t)(void*)&BhS[(nt + 1) & 1][0]);
      tdm_load_tile(Clo + (size_t)(nt + 1) * 16 * KM_D,
                    (unsigned)(size_t)(void*)&BlS[(nt + 1) & 1][0]);
    }
    const __bf16* chp = &BhS[nt & 1][col * KM_D];
    const __bf16* clp = &BlS[nt & 1][col * KM_D];
#else
    const __bf16* chp = Chi + (size_t)cl * KM_D;
    const __bf16* clp = Clo + (size_t)cl * KM_D;
#endif

    // ks-level software pipeline: preload B fragments for ks+1 while the
    // WMMAs of ks execute (kills the per-WMMA s_wait_dscnt stall).
    BU Bh[2], Bl[2];
    {
      const int bk = half * 8;
      Bh[0].h[0] = *(const v8bf*)(chp + bk);
      Bh[0].h[1] = *(const v8bf*)(chp + bk + 16);
      Bl[0].h[0] = *(const v8bf*)(clp + bk);
      Bl[0].h[1] = *(const v8bf*)(clp + bk + 16);
    }
    v8f acc0 = {}, acc1 = {};  // two independent chains -> 2x WMMA ILP per wave
#pragma unroll
    for (int ks = 0; ks < 4; ++ks) {
      const int cur = ks & 1, nxt = cur ^ 1;
      if (ks < 3) {
        const int bk = (ks + 1) * 32 + half * 8;
        Bh[nxt].h[0] = *(const v8bf*)(chp + bk);
        Bh[nxt].h[1] = *(const v8bf*)(chp + bk + 16);
        Bl[nxt].h[0] = *(const v8bf*)(clp + bk);
        Bl[nxt].h[1] = *(const v8bf*)(clp + bk + 16);
      }
      if (ks & 1) {
        acc1 = __builtin_amdgcn_wmma_f32_16x16x32_bf16(false, Ah[ks], false,
                                                       Bh[cur].v, (short)0, acc1,
                                                       false, false);
        acc0 = __builtin_amdgcn_wmma_f32_16x16x32_bf16(false, Ah[ks], false,
                                                       Bl[cur].v, (short)0, acc0,
                                                       false, false);
        acc1 = __builtin_amdgcn_wmma_f32_16x16x32_bf16(false, Al[ks], false,
                                                       Bh[cur].v, (short)0, acc1,
                                                       false, false);
      } else {
        acc0 = __builtin_amdgcn_wmma_f32_16x16x32_bf16(false, Ah[ks], false,
                                                       Bh[cur].v, (short)0, acc0,
                                                       false, false);
        acc1 = __builtin_amdgcn_wmma_f32_16x16x32_bf16(false, Ah[ks], false,
                                                       Bl[cur].v, (short)0, acc1,
                                                       false, false);
        acc0 = __builtin_amdgcn_wmma_f32_16x16x32_bf16(false, Al[ks], false,
                                                       Bh[cur].v, (short)0, acc0,
                                                       false, false);
      }
    }
    const float csh = 0.5f * csq[cl];
#pragma unroll
    for (int v = 0; v < 8; ++v) {
      float e = csh - (acc0[v] + acc1[v]);
      if (e < beste[v]) { beste[v] = e; bestn[v] = cl; }  // strict <: first-min wins
    }
  }

  // ---- Cross-lane argmin within each 16-lane half (ties -> smaller index) ----
#pragma unroll
  for (int v = 0; v < 8; ++v) {
    float e = beste[v];
    int   n = bestn[v];
#pragma unroll
    for (int m = 1; m <= 8; m <<= 1) {
      float oe = __shfl_xor(e, m, 32);
      int   on = __shfl_xor(n, m, 32);
      if (oe < e || (oe == e && on < n)) { e = oe; n = on; }
    }
    beste[v] = e;
    bestn[v] = n;
  }

  // ---- Labels, counts, inertia partial (lanes 0 / 16 own the two halves) ----
  float isum = 0.f;
  if (col == 0) {
#pragma unroll
    for (int v = 0; v < 8; ++v) {
      const int m = rowbase + half * 8 + v;
      labels[m] = bestn[v];                         // int32 bits
      isum += fmaxf(xs[v] + 2.0f * beste[v], 0.f);  // recovered d^2, clamped
      atomicAdd(counts + bestn[v], 1);              // integer: deterministic
    }
  }
  isum += __shfl_xor(isum, 16, 32);
  __shared__ float wpart[8];
  if (lane == 0) wpart[wv] = isum;
  __syncthreads();
  if (threadIdx.x == 0) {
    float t = 0.f;
#pragma unroll
    for (int w = 0; w < 8; ++w) t += wpart[w];  // fixed order
    iblk[blockIdx.x] = t;
  }
}

// ---------------------------------------------------------------------------
// Exclusive prefix sum of counts[512] -> offsets[512]. One block, 512 threads.
// ---------------------------------------------------------------------------
__global__ __launch_bounds__(512) void kmeans_scan(const int* __restrict__ counts,
                                                   int* __restrict__ offsets) {
  __shared__ int tmp[KM_K];
  const int t = threadIdx.x;
  const int c = counts[t];
  tmp[t] = c;
  __syncthreads();
  for (int d = 1; d < KM_K; d <<= 1) {
    const int add = (t >= d) ? tmp[t - d] : 0;
    __syncthreads();
    tmp[t] += add;
    __syncthreads();
  }
  offsets[t] = tmp[t] - c;  // exclusive
}

// ---------------------------------------------------------------------------
// Counting-sort scatter: order[] lists sample indices grouped by cluster.
// ---------------------------------------------------------------------------
__global__ __launch_bounds__(256) void kmeans_scatter(
    const int* __restrict__ labels, const int* __restrict__ offsets,
    int* __restrict__ cursor, int* __restrict__ order) {
  const int i = blockIdx.x * 256 + threadIdx.x;
  if (i >= KM_N) return;
  const int lbl = labels[i];
  const int pos = atomicAdd(cursor + lbl, 1);
  order[offsets[lbl] + pos] = i;
}

// ---------------------------------------------------------------------------
// Segmented mean: one block per cluster, thread t owns feature t (coalesced
// 512B row gathers). Row indices staged through LDS. Empty -> keep old.
// ---------------------------------------------------------------------------
__global__ __launch_bounds__(128) void kmeans_update(
    const float* __restrict__ X, const int* __restrict__ order,
    const int* __restrict__ offsets, const int* __restrict__ counts,
    float* __restrict__ cent, __bf16* __restrict__ Chi, __bf16* __restrict__ Clo,
    float* __restrict__ csq, float* __restrict__ outCent, int writeOut) {
  const int k = blockIdx.x;
  const int f = threadIdx.x;
  const int beg = offsets[k];
  const int cnt = counts[k];

  __shared__ int rows[128];
  float acc = 0.f;
  for (int base = 0; base < cnt; base += 128) {
    const int m = (cnt - base < 128) ? (cnt - base) : 128;
    if (f < m) rows[f] = order[beg + base + f];
    __syncthreads();
#pragma unroll 4
    for (int j = 0; j < m; ++j)
      acc += __builtin_nontemporal_load(X + (size_t)rows[j] * KM_D + f);
    __syncthreads();
  }

  float c = (cnt > 0) ? (acc / (float)cnt) : cent[(size_t)k * KM_D + f];
  finish_centroid(k, f, c, cent, Chi, Clo, csq, outCent, writeOut);
}

// ---------------------------------------------------------------------------
// Deterministic inertia reduction over block partials. One block.
// ---------------------------------------------------------------------------
__global__ __launch_bounds__(256) void kmeans_inertia(const float* __restrict__ iblk,
                                                      int n, float* __restrict__ out) {
  float s = 0.f;
  for (int i = threadIdx.x; i < n; i += 256) s += iblk[i];
#pragma unroll
  for (int m = 16; m >= 1; m >>= 1) s += __shfl_xor(s, m, 32);
  __shared__ float wsum[8];
  if ((threadIdx.x & 31) == 0) wsum[threadIdx.x >> 5] = s;
  __syncthreads();
  if (threadIdx.x == 0) {
    float t = 0.f;
#pragma unroll
    for (int w = 0; w < 8; ++w) t += wsum[w];
    *out = t;
  }
}

// ---------------------------------------------------------------------------
extern "C" void kernel_launch(void* const* d_in, const int* in_sizes, int n_in,
                              void* d_out, int out_size, void* d_ws, size_t ws_size,
                              hipStream_t stream) {
  (void)in_sizes; (void)n_in; (void)out_size; (void)ws_size;
  const float* X     = (const float*)d_in[0];
  const float* initC = (const float*)d_in[1];

  float* out     = (float*)d_out;
  float* outCent = out;                        // [K*D] f32
  int*   labels  = (int*)(out + KM_K * KM_D);  // [N] int32 bit patterns
  float* inertia = out + KM_K * KM_D + KM_N;   // [1] f32

  char* w = (char*)d_ws;
  auto carve = [&](size_t bytes) -> char* {
    char* p = w;
    w += (bytes + 255) & ~(size_t)255;
    return p;
  };
  float*  cent    = (float*)carve((size_t)KM_K * KM_D * 4);
  float*  csq     = (float*)carve((size_t)KM_K * 4);
  float*  xsq     = (float*)carve((size_t)KM_N * 4);
  __bf16* Chi     = (__bf16*)carve((size_t)KM_K * KM_D * 2);
  __bf16* Clo     = (__bf16*)carve((size_t)KM_K * KM_D * 2);
  int*    counts  = (int*)carve((size_t)KM_K * 4);
  int*    cursor  = (int*)carve((size_t)KM_K * 4);
  int*    offsets = (int*)carve((size_t)KM_K * 4);
  int*    order   = (int*)carve((size_t)KM_N * 4);
  float*  iblk    = (float*)carve((size_t)(KM_N / 128) * 4);

  const int NBLK = KM_N / 128;  // 3125 assign blocks

  kmeans_xsq<<<KM_N / 8, 256, 0, stream>>>(X, xsq);
  kmeans_centinit<<<KM_K, 128, 0, stream>>>(initC, cent, Chi, Clo, csq);

  for (int it = 0; it < KM_ITERS; ++it) {
    kmeans_zero<<<(KM_K + 255) / 256, 256, 0, stream>>>(counts, cursor);
    kmeans_assign<<<NBLK, 256, 0, stream>>>(X, xsq, Chi, Clo, csq, labels, counts,
                                            iblk);
    kmeans_scan<<<1, KM_K, 0, stream>>>(counts, offsets);
    kmeans_scatter<<<(KM_N + 255) / 256, 256, 0, stream>>>(labels, offsets, cursor,
                                                           order);
    kmeans_update<<<KM_K, 128, 0, stream>>>(X, order, offsets, counts, cent, Chi,
                                            Clo, csq, outCent,
                                            (it == KM_ITERS - 1) ? 1 : 0);
    kmeans_inertia<<<1, 256, 0, stream>>>(iblk, NBLK, inertia);
  }
}
